// E3NN_PhaseNet_simple_lengthless_54692113547908
// MI455X (gfx1250) — matrix operand order repaired and checked by reading, not compiled, not run on hardware
//
#include <hip/hip_runtime.h>

// ---------------------------------------------------------------------------
// E3NN PhaseNet on MI455X (gfx1250, wave32).
//   k_cg      : analytic real-basis Clebsch-Gordan tensors (unit Frobenius
//               norm, largest-|entry| positive — matches reference).
//   k_layer1  : (4x l=0) -> [(8,4),(8,6)]   thread/edge, atomic scatter
//   k_layer2  : [(8,4),(8,6)] -> same       wave/edge; CG in LDS; y staged in
//               LDS pre-arranged as f16 WMMA B-fragments; channel mix via
//               v_wmma_f32_16x16x32_f16 (K = 4 paths x 8 channels = 32)
//   k_layer3  : -> (4x l=0)                 thread/edge
//   k_head    : 4 -> 64 -> silu -> 4        thread/node
// Workspace (floats): [cg 11148 |pad 11264| h1 N*176 | h2 N*176 | h3 N*4]
// ---------------------------------------------------------------------------

typedef _Float16 v16h __attribute__((ext_vector_type(16)));
typedef float    v8f  __attribute__((ext_vector_type(8)));

#define HDIM 176
#define CG_L2_BASE 250
#define CG_L2_TOTAL 10648

struct W8 { const float* p[8]; };

// ------------------------- spherical harmonics -----------------------------
__device__ __forceinline__ void sh_one(int l, float z, float s,
                                       const float* cm, const float* sm, float* out) {
  const float F[13] = {1.f,1.f,2.f,6.f,24.f,120.f,720.f,5040.f,40320.f,
                       362880.f,3628800.f,39916800.f,479001600.f};
  float P[7];
  for (int m = 0; m <= l; ++m) {
    float coeff = 1.f;
    for (int k = 1; k <= m; ++k) coeff *= -(float)(2 * k - 1);
    float sp = 1.f;
    for (int k = 0; k < m; ++k) sp *= s;
    float p_prev = coeff * sp;
    if (l == m) { P[m] = p_prev; continue; }
    float p_curr = z * (float)(2 * m + 1) * p_prev;
    for (int ll = m + 2; ll <= l; ++ll) {
      float nxt = ((float)(2 * ll - 1) * z * p_curr - (float)(ll - 1 + m) * p_prev)
                  / (float)(ll - m);
      p_prev = p_curr; p_curr = nxt;
    }
    P[m] = p_curr;
  }
  for (int m = 0; m <= l; ++m) {
    float N = sqrtf((float)(2 * l + 1) * F[l - m] / F[l + m]);
    if (m == 0) out[l] = N * P[0];
    else {
      float q = 1.41421356237309515f * N * P[m];
      out[l + m] = q * cm[m];
      out[l - m] = q * sm[m];
    }
  }
}

__device__ __forceinline__ void compute_sh(float vx, float vy, float vz,
                                           float* sh4, float* sh6) {
  float r = sqrtf(vx * vx + vy * vy + vz * vz + 1e-12f);
  float x = vx / r, y = vy / r, z = vz / r;
  float s = sqrtf(x * x + y * y + 1e-12f);
  float c1 = x / s, s1 = y / s;
  float cm[7], sm[7];
  cm[0] = 1.f; sm[0] = 0.f;
  for (int m = 1; m < 7; ++m) {           // Chebyshev recurrence, no trig
    cm[m] = cm[m - 1] * c1 - sm[m - 1] * s1;
    sm[m] = sm[m - 1] * c1 + cm[m - 1] * s1;
  }
  sh_one(4, z, s, cm, sm, sh4);
  sh_one(6, z, s, cm, sm, sh6);
}

// ------------------------- CG tensors on device ----------------------------
__device__ __forceinline__ double dfact(int n) {
  const double F[20] = {1.,1.,2.,6.,24.,120.,720.,5040.,40320.,362880.,
    3628800.,39916800.,479001600.,6227020800.,87178291200.,1307674368000.,
    20922789888000.,355687428096000.,6402373705728000.,121645100408832000.};
  return F[n];
}

__device__ double wigner3j(int j1, int j2, int j3, int m1, int m2, int m3) {
  if (m1 + m2 + m3 != 0) return 0.0;
  if (abs(m1) > j1 || abs(m2) > j2 || abs(m3) > j3) return 0.0;
  double delta = dfact(j1 + j2 - j3) * dfact(j1 - j2 + j3) * dfact(-j1 + j2 + j3)
               / dfact(j1 + j2 + j3 + 1);
  double pre = sqrt(delta * dfact(j1 + m1) * dfact(j1 - m1) * dfact(j2 + m2)
                          * dfact(j2 - m2) * dfact(j3 + m3) * dfact(j3 - m3));
  int kmin = 0;
  if (j2 - j3 - m1 > kmin) kmin = j2 - j3 - m1;
  if (j1 - j3 + m2 > kmin) kmin = j1 - j3 + m2;
  int kmax = j1 + j2 - j3;
  if (j1 - m1 < kmax) kmax = j1 - m1;
  if (j2 + m2 < kmax) kmax = j2 + m2;
  double sum = 0.0;
  for (int k = kmin; k <= kmax; ++k) {
    double den = dfact(k) * dfact(j1 + j2 - j3 - k) * dfact(j1 - m1 - k)
               * dfact(j2 + m2 - k) * dfact(j3 - j2 + m1 + k) * dfact(j3 - j1 - m2 + k);
    sum += ((k & 1) ? -1.0 : 1.0) / den;
  }
  int ph = j1 - j2 - m3;
  double sgn = (ph % 2 != 0) ? -1.0 : 1.0;
  return sgn * pre * sum;
}

struct BC { int m; double re, im; };
// complex->real-SH basis change B_{m,mu} matching reference's _real_sh convention
__device__ __forceinline__ int bcoef(int l, int i, BC out[2]) {
  const double RS2 = 0.70710678118654752;
  int mu = i - l;
  if (mu == 0) { out[0] = {0, 1.0, 0.0}; return 1; }
  if (mu > 0) {
    int m = mu; double par = (m & 1) ? -RS2 : RS2;
    out[0] = {m, RS2, 0.0}; out[1] = {-m, par, 0.0}; return 2;
  }
  int m = -mu; double par = (m & 1) ? RS2 : -RS2;   // -(-1)^m / sqrt2 imaginary
  out[0] = {m, 0.0, RS2}; out[1] = {-m, 0.0, par}; return 2;
}

__global__ __launch_bounds__(256) void k_cg(float* cg) {
  const int L1t[12] = {0,0,4,4,4,4,6,6,6,6,4,6};
  const int L2t[12] = {4,6,4,4,6,6,4,4,6,6,4,6};
  const int L3t[12] = {4,6,4,6,4,6,4,6,4,6,0,0};
  const int OFF[12] = {0,81,250,979,2032,3085,4606,5659,7180,8701,10898,10979};
  int key = blockIdx.x;
  int l1 = L1t[key], l2 = L2t[key], l3 = L3t[key];
  int d1 = 2*l1+1, d2 = 2*l2+1, d3 = 2*l3+1;
  int n = d1 * d2 * d3;
  float* out = cg + OFF[key];
  int tid = threadIdx.x;

  float vals[9];
  double bestabs = -1.0; int bestidx = 0x7fffffff;
  int t = 0;
  for (int idx = tid; idx < n; idx += 256, ++t) {
    int i1 = idx / (d2 * d3);
    int rem = idx - i1 * d2 * d3;
    int i2 = rem / d3, i3 = rem - i2 * d3;
    BC A[2], B[2], C[2];
    int na = bcoef(l1, i1, A), nb = bcoef(l2, i2, B), nc = bcoef(l3, i3, C);
    double re = 0.0;
    for (int a = 0; a < na; ++a)
      for (int b = 0; b < nb; ++b)
        for (int c = 0; c < nc; ++c) {
          double w = wigner3j(l1, l2, l3, A[a].m, B[b].m, C[c].m);
          if (w == 0.0) continue;
          double r12 = A[a].re * B[b].re - A[a].im * B[b].im;
          double i12 = A[a].re * B[b].im + A[a].im * B[b].re;
          double rr  = r12 * C[c].re - i12 * C[c].im;
          re += w * rr;
        }
    vals[t] = (float)re;
    double aa = fabs(re);
    if (aa > bestabs) { bestabs = aa; bestidx = idx; }
  }
  __shared__ double sa[256]; __shared__ int si[256]; __shared__ float ssign;
  sa[tid] = bestabs; si[tid] = bestidx;
  __syncthreads();
  for (int off = 128; off; off >>= 1) {
    if (tid < off) {
      if (sa[tid + off] > sa[tid] ||
          (sa[tid + off] == sa[tid] && si[tid + off] < si[tid])) {
        sa[tid] = sa[tid + off]; si[tid] = si[tid + off];
      }
    }
    __syncthreads();
  }
  if (tid == (si[0] & 255)) {
    float v = vals[si[0] >> 8];
    ssign = (v < 0.f) ? -1.f : 1.f;
  }
  __syncthreads();
  float sgn = ssign;
  t = 0;
  for (int idx = tid; idx < n; idx += 256, ++t) out[idx] = vals[t] * sgn;
}

// ------------------------------- layer 1 -----------------------------------
__global__ __launch_bounds__(256) void k_layer1(const int* xspec, const int* ei,
    const float* evec, const float* emb, const float* w044, const float* w066,
    const float* cg, float* h1, int NE) {
  int e = blockIdx.x * blockDim.x + threadIdx.x;
  if (e >= NE) return;
  int src = ei[e], dst = ei[NE + e];
  float sh4[9], sh6[13];
  compute_sh(evec[3*e], evec[3*e+1], evec[3*e+2], sh4, sh6);
  int sp = xspec[src];
  float h[4];
  #pragma unroll
  for (int u = 0; u < 4; ++u) h[u] = emb[sp * 4 + u];
  float xw4[8], xw6[8];
  #pragma unroll
  for (int v = 0; v < 8; ++v) {
    float a4 = 0.f, a6 = 0.f;
    #pragma unroll
    for (int u = 0; u < 4; ++u) { a4 += h[u]*w044[u*8+v]; a6 += h[u]*w066[u*8+v]; }
    xw4[v] = a4; xw6[v] = a6;
  }
  const float* cg044 = cg;          // (1,9,9)
  const float* cg066 = cg + 81;     // (1,13,13)
  float csh4[9], csh6[13];
  #pragma unroll
  for (int k = 0; k < 9; ++k) {
    float a = 0.f;
    #pragma unroll
    for (int j = 0; j < 9; ++j) a += sh4[j] * cg044[j*9+k];
    csh4[k] = a;
  }
  #pragma unroll
  for (int k = 0; k < 13; ++k) {
    float a = 0.f;
    #pragma unroll
    for (int j = 0; j < 13; ++j) a += sh6[j] * cg066[j*13+k];
    csh6[k] = a;
  }
  const float S1 = 0.14433756729740643f;  // (1/sqrt(4)) * (1/sqrt(12))
  float* o = h1 + (size_t)dst * HDIM;
  #pragma unroll
  for (int v = 0; v < 8; ++v) {
    #pragma unroll
    for (int k = 0; k < 9; ++k)  atomicAdd(o + v*9 + k,       xw4[v]*csh4[k]*S1);
    #pragma unroll
    for (int k = 0; k < 13; ++k) atomicAdd(o + 72 + v*13 + k, xw6[v]*csh6[k]*S1);
  }
}

// ------------------------------- layer 2 -----------------------------------
// Per-wave LDS scratch:
//   s_xsh[wv]  : x (176 f32) + sh (22 f32)
//   s_csh[wv]  : padded csh, l_out=4 paths at pl*117, l_out=6 at 468+pl*169
//   s_yT[wv]   : y in f16, pre-arranged as WMMA B fragments:
//                [G][n][kk] at G*512 + n*32 + kk, zero-padded for n>=dk.
__global__ __launch_bounds__(256) void k_layer2(const int* ei, const float* evec,
    const float* hin, W8 w, const float* cgl2, float* hout, int NE) {
  __shared__ float s_cg[CG_L2_TOTAL];                           // 42.6 KB
  __shared__ float s_xsh[8][200];
  __shared__ float s_csh[8][1152];
  __shared__ _Float16 s_yT[8][1024] __attribute__((aligned(32)));
  const int tid = threadIdx.x;
  const int lane = tid & 31, wv = tid >> 5;

  for (int i = tid; i < CG_L2_TOTAL; i += 256) s_cg[i] = cgl2[i];

  // Edge-invariant A fragments (packed path weights, f16).
  // A 16x32 f16, contiguous k order used here (B matches): lane = m + 16*(k>>4); elem e -> k = g*16 + e
  // => pl = k>>3 = 2*g + (e>>3) ; u = k&7 = e&7
  v16h a4, a6;
  {
    int m = lane & 15, g = lane >> 4;
    #pragma unroll
    for (int e2 = 0; e2 < 16; ++e2) {
      int hi = e2 >> 3;                 // 0/1 (compile-time after unroll)
      int u  = e2 & 7;
      const float* wp4 = g ? (hi ? w.p[6] : w.p[4]) : (hi ? w.p[2] : w.p[0]);
      const float* wp6 = g ? (hi ? w.p[7] : w.p[5]) : (hi ? w.p[3] : w.p[1]);
      float w4v = 0.f, w6v = 0.f;
      if (m < 8) { w4v = wp4[u*8 + m]; w6v = wp6[u*8 + m]; }
      a4[e2] = (_Float16)w4v; a6[e2] = (_Float16)w6v;
    }
  }
  __syncthreads();

  float* xs  = s_xsh[wv];
  float* shs = xs + 176;
  float* csh = s_csh[wv];
  _Float16* yT = s_yT[wv];
  const float S2 = 0.051031036307982884f;  // 1/sqrt(32)/sqrt(12)

  int wave  = blockIdx.x * 8 + wv;
  int wstep = gridDim.x * 8;
  for (int e = wave; e < NE; e += wstep) {       // uniform per wave (EXEC all-1 at WMMA)
    int src = ei[e], dst = ei[NE + e];
    for (int t = lane; t < HDIM; t += 32) xs[t] = hin[(size_t)src * HDIM + t];
    float sh4[9], sh6[13];
    compute_sh(evec[3*e], evec[3*e+1], evec[3*e+2], sh4, sh6);
    if (lane < 9)  shs[lane] = sh4[lane];
    if (lane < 13) shs[9 + lane] = sh6[lane];
    asm volatile("s_wait_dscnt 0" ::: "memory");

    // csh_p[i,k] = sum_j sh[j] * CG_p[i,j,k]   (padded affine layout)
    // path p: g6 = p&1, pl = p>>1 ; dst off = g6 ? 468+pl*169 : pl*117
    {
      constexpr int DI[8]  = {9,9,9,9,13,13,13,13};
      constexpr int DJ[8]  = {9,9,13,13,9,9,13,13};
      constexpr int DK[8]  = {9,13,9,13,9,13,9,13};
      constexpr int SHO[8] = {0,0,9,9,0,0,9,9};
      constexpr int CGO[8] = {0,729,1782,2835,4356,5409,6930,8451};
      constexpr int CSO[8] = {0,468,117,637,234,806,351,975};
      #pragma unroll
      for (int p = 0; p < 8; ++p) {
        const int di = DI[p], dj = DJ[p], dk = DK[p];
        const float* cgp = s_cg + CGO[p];
        const float* shp = shs + SHO[p];
        float* cdst = csh + CSO[p];
        const int nn = di * dk;
        for (int idx = lane; idx < nn; idx += 32) {
          int i = idx / dk, k = idx - i * dk;
          float acc = 0.f;
          for (int j = 0; j < dj; ++j) acc += shp[j] * cgp[(i*dj + j)*dk + k];
          cdst[idx] = acc;
        }
      }
    }
    asm volatile("s_wait_dscnt 0" ::: "memory");

    // y_p[u,k] = sum_i x[u,i] * csh_p[i,k] ; one K-row (kk) per lane, written
    // transposed + zero-padded in f16 => B fragment is a contiguous v16h.
    {
      int kk = lane, pl = kk >> 3, u = kk & 7;
      int di = (pl < 2) ? 9 : 13;
      int xo = (pl < 2) ? 0 : 72;
      const float* xsrc = xs + xo + u * di;
      const float* c4 = csh + pl * 117;          // dk = 9
      const float* c6 = csh + 468 + pl * 169;    // dk = 13
      #pragma unroll
      for (int n = 0; n < 16; ++n) {
        float a4v = 0.f, a6v = 0.f;
        if (n < 9)  for (int i = 0; i < di; ++i) a4v += xsrc[i] * c4[i*9  + n];
        if (n < 13) for (int i = 0; i < di; ++i) a6v += xsrc[i] * c6[i*13 + n];
        yT[n*32 + kk]       = (_Float16)a4v;
        yT[512 + n*32 + kk] = (_Float16)a6v;
      }
    }
    asm volatile("s_wait_dscnt 0" ::: "memory");

    // Channel mix: out[v,k] = sum_{p,u} w_p[u,v] * y_p[u,k]  (K=32) via WMMA.
    // B 32x16 f16, same contiguous k order as A: lane = n + 16*(k>=16); elem e -> k = (lane>>4)*16 + e
    {
      int n = lane & 15, kb = lane >> 4;
      v16h b4 = *(const v16h*)(yT + n*32 + kb*16);
      v16h b6 = *(const v16h*)(yT + 512 + n*32 + kb*16);
      v8f z = {0.f,0.f,0.f,0.f,0.f,0.f,0.f,0.f};
      v8f d4 = __builtin_amdgcn_wmma_f32_16x16x32_f16(false, a4, false, b4,
                                                      (short)0, z, false, false);
      v8f d6 = __builtin_amdgcn_wmma_f32_16x16x32_f16(false, a6, false, b6,
                                                      (short)0, z, false, false);
      float* o = hout + (size_t)dst * HDIM;
      if (lane < 9) {
        #pragma unroll
        for (int v = 0; v < 8; ++v) atomicAdd(o + v*9 + lane, d4[v] * S2);
      }
      if (lane < 13) {
        #pragma unroll
        for (int v = 0; v < 8; ++v) atomicAdd(o + 72 + v*13 + lane, d6[v] * S2);
      }
    }
  }
}

// ------------------------------- layer 3 -----------------------------------
__global__ __launch_bounds__(256) void k_layer3(const int* ei, const float* evec,
    const float* h2, const float* w440, const float* w660, const float* cg,
    float* h3, int NE) {
  int e = blockIdx.x * blockDim.x + threadIdx.x;
  if (e >= NE) return;
  int src = ei[e], dst = ei[NE + e];
  float sh4[9], sh6[13];
  compute_sh(evec[3*e], evec[3*e+1], evec[3*e+2], sh4, sh6);
  const float* cg440 = cg + 10898;   // (9,9,1)
  const float* cg660 = cg + 10979;   // (13,13,1)
  float csh4[9], csh6[13];
  #pragma unroll
  for (int i = 0; i < 9; ++i) {
    float a = 0.f;
    #pragma unroll
    for (int j = 0; j < 9; ++j) a += sh4[j] * cg440[i*9 + j];
    csh4[i] = a;
  }
  #pragma unroll
  for (int i = 0; i < 13; ++i) {
    float a = 0.f;
    #pragma unroll
    for (int j = 0; j < 13; ++j) a += sh6[j] * cg660[i*13 + j];
    csh6[i] = a;
  }
  float out[4] = {0.f,0.f,0.f,0.f};
  const float* xx = h2 + (size_t)src * HDIM;
  #pragma unroll
  for (int u = 0; u < 8; ++u) {
    float t4 = 0.f, t6 = 0.f;
    #pragma unroll
    for (int i = 0; i < 9; ++i)  t4 += xx[u*9 + i] * csh4[i];
    #pragma unroll
    for (int i = 0; i < 13; ++i) t6 += xx[72 + u*13 + i] * csh6[i];
    #pragma unroll
    for (int v = 0; v < 4; ++v) out[v] += t4 * w440[u*4+v] + t6 * w660[u*4+v];
  }
  const float S3 = 0.0721687836487032f;  // (1/sqrt(16)) * (1/sqrt(12))
  #pragma unroll
  for (int v = 0; v < 4; ++v) atomicAdd(&h3[(size_t)dst*4 + v], out[v] * S3);
}

// --------------------------------- head ------------------------------------
__global__ __launch_bounds__(256) void k_head(const float* h3, const float* W1,
    const float* W2, float* out, int NN) {
  int nidx = blockIdx.x * blockDim.x + threadIdx.x;
  if (nidx >= NN) return;
  float a0 = h3[nidx*4+0], a1 = h3[nidx*4+1], a2 = h3[nidx*4+2], a3 = h3[nidx*4+3];
  float acc[4] = {0.f,0.f,0.f,0.f};
  for (int j = 0; j < 64; ++j) {
    float t = (a0*W1[j] + a1*W1[64+j] + a2*W1[128+j] + a3*W1[192+j]) * 0.5f;
    float sv = t / (1.f + expf(-t));   // silu
    #pragma unroll
    for (int v = 0; v < 4; ++v) acc[v] += sv * W2[j*4 + v];
  }
  #pragma unroll
  for (int v = 0; v < 4; ++v) out[nidx*4 + v] = acc[v] * 0.125f;
}

// ------------------------------- launcher ----------------------------------
extern "C" void kernel_launch(void* const* d_in, const int* in_sizes, int n_in,
                              void* d_out, int out_size, void* d_ws, size_t ws_size,
                              hipStream_t stream) {
  const int*   xspec = (const int*)d_in[0];
  const int*   ei    = (const int*)d_in[1];
  const float* evec  = (const float*)d_in[2];
  const float* emb   = (const float*)d_in[3];
  const float* w044  = (const float*)d_in[4];   // conv_weights flatten: sorted dict keys
  const float* w066  = (const float*)d_in[5];
  W8 wl2;
  for (int p = 0; p < 8; ++p) wl2.p[p] = (const float*)d_in[6 + p];
  const float* w440  = (const float*)d_in[14];
  const float* w660  = (const float*)d_in[15];
  const float* W1    = (const float*)d_in[16];
  const float* W2    = (const float*)d_in[17];

  const int NN = in_sizes[0];
  const int NE = in_sizes[1] / 2;

  float* ws = (float*)d_ws;
  float* cg = ws;                               // 11148 floats, pad to 11264
  float* h1 = ws + 11264;
  float* h2 = h1 + (size_t)NN * HDIM;
  float* h3 = h2 + (size_t)NN * HDIM;
  // total ws need: (11264 + NN*(2*176+4)) * 4 bytes  (~71.3 MB for NN=50000)

  hipMemsetAsync(h1, 0, sizeof(float) * ((size_t)NN * (2 * HDIM + 4)), stream);
  k_cg<<<12, 256, 0, stream>>>(cg);
  k_layer1<<<(NE + 255) / 256, 256, 0, stream>>>(xspec, ei, evec, emb, w044, w066,
                                                 cg, h1, NE);
  k_layer2<<<2048, 256, 0, stream>>>(ei, evec, h1, wl2, cg + CG_L2_BASE, h2, NE);
  k_layer3<<<(NE + 255) / 256, 256, 0, stream>>>(ei, evec, h2, w440, w660, cg, h3, NE);
  k_head<<<(NN + 255) / 256, 256, 0, stream>>>(h3, W1, W2, (float*)d_out, NN);
}
